// SEGNO_16870631538840
// MI455X (gfx1250) — matrix operand order repaired and checked by reading, not compiled, not run on hardware
//
#include <hip/hip_runtime.h>
#include <hip/hip_bf16.h>

typedef __attribute__((ext_vector_type(16))) _Float16 v16h;
typedef __attribute__((ext_vector_type(8)))  float    v8f;

#define HDIM 64
#define TILE 16
#define EWAVES 4

__device__ __forceinline__ float silu_f(float x) {
  return x / (1.0f + __expf(-x));
}

union AFragV { v16h v; uint4 q[2]; };

// A-matrix fragment (16x32 f16) for 16x16x32 WMMA, built from a row-major
// 16x64 f16 tile. ISA layout: lane L (row=L&15, hs=L>>4) needs two contiguous
// 8-half chunks: K in [k0+hs*8, +8) and [k0+16+hs*8, +8).
// LDS version (tile base = 16-row tile):
__device__ __forceinline__ v16h load_afrag(const _Float16* tile16x64, int lane, int k0) {
  const int row = lane & 15;
  const int hs  = lane >> 4;
  const uint4* p = (const uint4*)(tile16x64 + row * HDIM + k0 + hs * 8);
  AFragV f;
  f.q[0] = p[0];
  f.q[1] = p[2];   // +16 halfs
  return f.v;
}

// Global version: lane supplies its own row pointer (h16 + row*64).
__device__ __forceinline__ v16h load_afrag_g(const _Float16* rowp, int lane, int k0) {
  const int hs = lane >> 4;
  const uint4* p = (const uint4*)(rowp + k0 + hs * 8);
  AFragV f;
  f.q[0] = p[0];
  f.q[1] = p[2];   // +16 halfs
  return f.v;
}

// B-matrix fragment (32x16 f16) from transposed weights WT[n][k] in LDS.
// lane L: col n = ntile*16 + (L&15); K in [k0+(L>>4)*8, +8) and [k0+16+(L>>4)*8, +8).
__device__ __forceinline__ v16h load_bfrag(const _Float16* WT64x64, int lane, int ntile, int k0) {
  const int n  = ntile * 16 + (lane & 15);
  const int kb = k0 + ((lane >> 4) << 3);
  const uint4* p = (const uint4*)(WT64x64 + n * HDIM + kb);
  AFragV f;
  f.q[0] = p[0];
  f.q[1] = p[2];
  return f.v;
}

__device__ __forceinline__ v8f wmma16(v16h a, v16h b, v8f c) {
  return __builtin_amdgcn_wmma_f32_16x16x32_f16(false, a, false, b, (short)0, c, false, false);
}

struct EdgeLDS {
  _Float16 W1aT[HDIM * HDIM];   // e_W1 rows 0..63   (h[row] block), transposed [n][k]
  _Float16 W1bT[HDIM * HDIM];   // e_W1 rows 64..127 (h[col] block), transposed
  _Float16 W2T [HDIM * HDIM];   // e_W2 transposed
  _Float16 CW1T[HDIM * HDIM];   // c_W1 transposed
  float eb1[HDIM], eb2[HDIM], cb1[HDIM], cw2[HDIM];
  float w1r[HDIM], W1e0[HDIM], W1e1[HDIM];   // e_W1 rows 128 (radial), 129/130 (edge_attr)
  _Float16 stage[EWAVES][TILE * HDIM];       // inter-GEMM relayout buffer
  int   rowi[EWAVES][TILE];
  float cdx[EWAVES][TILE], cdy[EWAVES][TILE], cdz[EWAVES][TILE];
  float rad[EWAVES][TILE];
  float ea0[EWAVES][TILE], ea1[EWAVES][TILE];
};

// Fused edge pipeline: e-MLP (2 layers, SiLU) -> c-MLP -> scatter(agg_h, acc3).
// One wave32 per 16-edge tile; all GEMMs via v_wmma_f32_16x16x32_f16.
// A operands for GEMM1 gathered straight from the L2-resident f16 mirror of h.
__global__ __launch_bounds__(EWAVES * 32) void edge_kernel(
    const _Float16* __restrict__ h16, const float* __restrict__ x,
    const int* __restrict__ rowI, const int* __restrict__ colI,
    const float* __restrict__ eattr,
    const float* __restrict__ eW1, const float* __restrict__ eb1,
    const float* __restrict__ eW2, const float* __restrict__ eb2,
    const float* __restrict__ cW1, const float* __restrict__ cb1,
    const float* __restrict__ cW2,
    float* __restrict__ acc3, float* __restrict__ aggh, int E) {
  __shared__ EdgeLDS L;
  const int tid = threadIdx.x;
  for (int i = tid; i < HDIM * HDIM; i += blockDim.x) {
    const int k = i >> 6, n = i & 63;
    L.W1aT[n * HDIM + k] = (_Float16)eW1[k * HDIM + n];
    L.W1bT[n * HDIM + k] = (_Float16)eW1[(k + 64) * HDIM + n];
    L.W2T [n * HDIM + k] = (_Float16)eW2[i];
    L.CW1T[n * HDIM + k] = (_Float16)cW1[i];
  }
  for (int n = tid; n < HDIM; n += blockDim.x) {
    L.eb1[n] = eb1[n]; L.eb2[n] = eb2[n]; L.cb1[n] = cb1[n]; L.cw2[n] = cW2[n];
    L.w1r[n]  = eW1[128 * HDIM + n];
    L.W1e0[n] = eW1[129 * HDIM + n];
    L.W1e1[n] = eW1[130 * HDIM + n];
  }
  __syncthreads();

  const int wave = tid >> 5;
  const int lane = tid & 31;
  const int nlo  = lane & 15;
  const int hs   = lane >> 4;
  const int ntiles = (E + TILE - 1) / TILE;

  for (int t = blockIdx.x * EWAVES + wave; t < ntiles; t += gridDim.x * EWAVES) {
    const int e0 = t * TILE;
    // --- per-edge scalars (lanes 0..15); in-wave LDS ordering makes this safe ---
    if (lane < TILE) {
      const int e = e0 + lane;
      int r = 0; float dx = 0, dy = 0, dz = 0, a0 = 0, a1 = 0;
      if (e < E) {
        r = rowI[e];
        const int c = colI[e];
        dx = x[r * 3 + 0] - x[c * 3 + 0];
        dy = x[r * 3 + 1] - x[c * 3 + 1];
        dz = x[r * 3 + 2] - x[c * 3 + 2];
        a0 = eattr[e * 2 + 0]; a1 = eattr[e * 2 + 1];
      }
      L.rowi[wave][lane] = r;
      L.cdx[wave][lane] = dx; L.cdy[wave][lane] = dy; L.cdz[wave][lane] = dz;
      L.rad[wave][lane] = dx * dx + dy * dy + dz * dz;
      L.ea0[wave][lane] = a0; L.ea1[wave][lane] = a1;
    }
    // --- A operands straight from global f16 h (L2-resident), 2x b128 each ---
    const int eg = e0 + nlo;
    const int ec = (eg < E) ? eg : (E - 1);
    const _Float16* hr16 = h16 + (size_t)rowI[ec] * HDIM;
    const _Float16* hc16 = h16 + (size_t)colI[ec] * HDIM;
    v16h ar0 = load_afrag_g(hr16, lane, 0);
    v16h ar1 = load_afrag_g(hr16, lane, 32);
    v16h ac0 = load_afrag_g(hc16, lane, 0);
    v16h ac1 = load_afrag_g(hc16, lane, 32);

    // hoist per-row epilogue scalars into registers (after the LDS writes above)
    float radm[8], a0m[8], a1m[8];
    int   rn[8];
#pragma unroll
    for (int r8 = 0; r8 < 8; ++r8) {
      const int m = r8 + hs * 8;
      radm[r8] = L.rad[wave][m];
      a0m[r8]  = L.ea0[wave][m];
      a1m[r8]  = L.ea1[wave][m];
      rn[r8]   = L.rowi[wave][m];
    }

    // ---------------- GEMM1: m1 = e_in @ e_W1  (K=64+64 via two A blocks) ----
    v8f acc[4];
#pragma unroll
    for (int nt = 0; nt < 4; ++nt) {
      v8f cc = {};
      cc = wmma16(ar0, load_bfrag(L.W1aT, lane, nt, 0),  cc);
      cc = wmma16(ar1, load_bfrag(L.W1aT, lane, nt, 32), cc);
      cc = wmma16(ac0, load_bfrag(L.W1bT, lane, nt, 0),  cc);
      cc = wmma16(ac1, load_bfrag(L.W1bT, lane, nt, 32), cc);
      acc[nt] = cc;
    }
    // epilogue 1: + bias + radial*w1r + edge_attr@W1e ; SiLU ; stage as f16
#pragma unroll
    for (int nt = 0; nt < 4; ++nt) {
      const int n = nt * 16 + nlo;
      const float bn = L.eb1[n], wr = L.w1r[n], w0 = L.W1e0[n], w1 = L.W1e1[n];
#pragma unroll
      for (int r8 = 0; r8 < 8; ++r8) {
        const int m = r8 + hs * 8;
        const float val = acc[nt][r8] + bn + radm[r8] * wr + a0m[r8] * w0 + a1m[r8] * w1;
        L.stage[wave][m * HDIM + n] = (_Float16)silu_f(val);
      }
    }
    // ---------------- GEMM2: m = silu(m1 @ e_W2 + b) ------------------------
    v16h a20 = load_afrag(L.stage[wave], lane, 0);
    v16h a21 = load_afrag(L.stage[wave], lane, 32);
#pragma unroll
    for (int nt = 0; nt < 4; ++nt) {
      v8f cc = {};
      cc = wmma16(a20, load_bfrag(L.W2T, lane, nt, 0),  cc);
      cc = wmma16(a21, load_bfrag(L.W2T, lane, nt, 32), cc);
      acc[nt] = cc;
    }
    // epilogue 2: SiLU, scatter segment_sum(m) -> agg_h[row], restage f16
#pragma unroll
    for (int nt = 0; nt < 4; ++nt) {
      const int n = nt * 16 + nlo;
      const float bn = L.eb2[n];
#pragma unroll
      for (int r8 = 0; r8 < 8; ++r8) {
        const int m = r8 + hs * 8;
        const float val = silu_f(acc[nt][r8] + bn);
        if (e0 + m < E)
          atomicAdd(&aggh[(size_t)rn[r8] * HDIM + n], val);
        L.stage[wave][m * HDIM + n] = (_Float16)val;
      }
    }
    // ---------------- GEMM3: w = silu(m @ c_W1 + b) @ c_W2 ------------------
    v16h a30 = load_afrag(L.stage[wave], lane, 0);
    v16h a31 = load_afrag(L.stage[wave], lane, 32);
    float p[8];
#pragma unroll
    for (int r8 = 0; r8 < 8; ++r8) p[r8] = 0.0f;
#pragma unroll
    for (int nt = 0; nt < 4; ++nt) {
      v8f cc = {};
      cc = wmma16(a30, load_bfrag(L.CW1T, lane, nt, 0),  cc);
      cc = wmma16(a31, load_bfrag(L.CW1T, lane, nt, 32), cc);
      const int n = nt * 16 + nlo;
      const float cb = L.cb1[n], cw = L.cw2[n];
#pragma unroll
      for (int r8 = 0; r8 < 8; ++r8) p[r8] += silu_f(cc[r8] + cb) * cw;
    }
    // reduce the c_W2 dot over the 16 lanes of each half-wave
#pragma unroll
    for (int off = 1; off < 16; off <<= 1) {
#pragma unroll
      for (int r8 = 0; r8 < 8; ++r8) p[r8] += __shfl_xor(p[r8], off, 32);
    }
    if (nlo == 0) {  // lanes 0 and 16: scatter trans = coord_diff * w
#pragma unroll
      for (int r8 = 0; r8 < 8; ++r8) {
        const int m = r8 + hs * 8;
        if (e0 + m < E) {
          const float w = p[r8];
          atomicAdd(&acc3[(size_t)rn[r8] * 3 + 0], L.cdx[wave][m] * w);
          atomicAdd(&acc3[(size_t)rn[r8] * 3 + 1], L.cdy[wave][m] * w);
          atomicAdd(&acc3[(size_t)rn[r8] * 3 + 2], L.cdz[wave][m] * w);
        }
      }
    }
  }
}

// Velocity gate MLP + second-order update + speed renorm + coord update.
__global__ __launch_bounds__(256) void vel_kernel(
    const float* __restrict__ h, const float* __restrict__ acc3,
    const float* __restrict__ cnt, const float* __restrict__ sp0,
    float* __restrict__ x, float* __restrict__ v,
    const float* __restrict__ vW1, const float* __restrict__ vb1,
    const float* __restrict__ vW2, const float* __restrict__ vb2, int N) {
  __shared__ float W1[HDIM * HDIM];
  __shared__ float B1[HDIM], W2[HDIM];
  for (int i = threadIdx.x; i < HDIM * HDIM; i += blockDim.x) W1[i] = vW1[i];
  for (int i = threadIdx.x; i < HDIM; i += blockDim.x) { B1[i] = vb1[i]; W2[i] = vW2[i]; }
  __syncthreads();
  const float b2 = vb2[0];
  const float inv7 = 1.0f / 7.0f;
  for (int i = blockIdx.x * blockDim.x + threadIdx.x; i < N; i += gridDim.x * blockDim.x) {
    float hr[HDIM];
    const float4* hp = (const float4*)(h + (size_t)i * HDIM);
#pragma unroll
    for (int q = 0; q < 16; ++q) {
      float4 f = hp[q];
      hr[q * 4 + 0] = f.x; hr[q * 4 + 1] = f.y; hr[q * 4 + 2] = f.z; hr[q * 4 + 3] = f.w;
    }
    float vs = b2;
    for (int j = 0; j < HDIM; ++j) {
      float s = B1[j];
#pragma unroll
      for (int k = 0; k < HDIM; ++k) s += hr[k] * W1[k * HDIM + j];
      vs += silu_f(s) * W2[j];
    }
    const float invc = 1.0f / fmaxf(cnt[i], 1.0f);
    float vx = vs * v[(size_t)i * 3 + 0] + acc3[(size_t)i * 3 + 0] * invc * inv7;
    float vy = vs * v[(size_t)i * 3 + 1] + acc3[(size_t)i * 3 + 1] * invc * inv7;
    float vz = vs * v[(size_t)i * 3 + 2] + acc3[(size_t)i * 3 + 2] * invc * inv7;
    const float sc = sp0[i] / (sqrtf(vx * vx + vy * vy + vz * vz) + 1e-8f);
    vx *= sc; vy *= sc; vz *= sc;
    v[(size_t)i * 3 + 0] = vx; v[(size_t)i * 3 + 1] = vy; v[(size_t)i * 3 + 2] = vz;
    x[(size_t)i * 3 + 0] += vx * inv7;
    x[(size_t)i * 3 + 1] += vy * inv7;
    x[(size_t)i * 3 + 2] += vz * inv7;
  }
}

// Node MLP with uniform residual: h_next = 2*h_in + MLP([h_in, agg]).
// 64 threads per node (feature-per-thread), 4 nodes per 256-thread block.
// Also refreshes the f16 mirror of h used by the edge kernel's WMMA gathers.
__global__ __launch_bounds__(256) void nodeh_kernel(
    float* __restrict__ h, _Float16* __restrict__ h16,
    const float* __restrict__ agg,
    const float* __restrict__ nW1, const float* __restrict__ nb1,
    const float* __restrict__ nW2, const float* __restrict__ nb2, int N) {
  __shared__ float W1[128 * HDIM];
  __shared__ float W2[HDIM * HDIM];
  __shared__ float B1[HDIM], B2[HDIM];
  __shared__ float hin[4][HDIM], ain[4][HDIM], hid[4][HDIM];
  for (int i = threadIdx.x; i < 128 * HDIM; i += blockDim.x) W1[i] = nW1[i];
  for (int i = threadIdx.x; i < HDIM * HDIM; i += blockDim.x) W2[i] = nW2[i];
  for (int i = threadIdx.x; i < HDIM; i += blockDim.x) { B1[i] = nb1[i]; B2[i] = nb2[i]; }
  __syncthreads();
  const int g = threadIdx.x >> 6;
  const int j = threadIdx.x & 63;
  for (int base = blockIdx.x * 4; base < N; base += gridDim.x * 4) {
    const int i = base + g;
    if (i < N) {
      hin[g][j] = h[(size_t)i * HDIM + j];
      ain[g][j] = agg[(size_t)i * HDIM + j];
    }
    __syncthreads();
    if (i < N) {
      float s = B1[j];
      for (int k = 0; k < HDIM; ++k) s += hin[g][k] * W1[k * HDIM + j];
      for (int k = 0; k < HDIM; ++k) s += ain[g][k] * W1[(k + 64) * HDIM + j];
      hid[g][j] = silu_f(s);
    }
    __syncthreads();
    if (i < N) {
      float o = B2[j];
      for (int k = 0; k < HDIM; ++k) o += hid[g][k] * W2[k * HDIM + j];
      const float hnew = 2.0f * hin[g][j] + o;
      h[(size_t)i * HDIM + j]   = hnew;
      h16[(size_t)i * HDIM + j] = (_Float16)hnew;
    }
    __syncthreads();
  }
}

__global__ __launch_bounds__(256) void embed_kernel(
    const float* __restrict__ his, const float* __restrict__ loc,
    const float* __restrict__ vel, const float* __restrict__ embW,
    const float* __restrict__ embB, float* __restrict__ x,
    float* __restrict__ h, _Float16* __restrict__ h16,
    float* __restrict__ v, float* __restrict__ sp0, int N) {
  __shared__ float W[8 * HDIM];
  __shared__ float B[HDIM];
  for (int i = threadIdx.x; i < 8 * HDIM; i += blockDim.x) W[i] = embW[i];
  for (int i = threadIdx.x; i < HDIM; i += blockDim.x) B[i] = embB[i];
  __syncthreads();
  for (int i = blockIdx.x * blockDim.x + threadIdx.x; i < N; i += gridDim.x * blockDim.x) {
    float in[8];
#pragma unroll
    for (int k = 0; k < 8; ++k) in[k] = his[(size_t)i * 8 + k];
    for (int j = 0; j < HDIM; ++j) {
      float s = B[j];
#pragma unroll
      for (int k = 0; k < 8; ++k) s += in[k] * W[k * HDIM + j];
      h[(size_t)i * HDIM + j]   = s;
      h16[(size_t)i * HDIM + j] = (_Float16)s;
    }
    const float vx = vel[(size_t)i * 3 + 0];
    const float vy = vel[(size_t)i * 3 + 1];
    const float vz = vel[(size_t)i * 3 + 2];
    x[(size_t)i * 3 + 0] = loc[(size_t)i * 3 + 0];
    x[(size_t)i * 3 + 1] = loc[(size_t)i * 3 + 1];
    x[(size_t)i * 3 + 2] = loc[(size_t)i * 3 + 2];
    v[(size_t)i * 3 + 0] = vx; v[(size_t)i * 3 + 1] = vy; v[(size_t)i * 3 + 2] = vz;
    sp0[i] = sqrtf(vx * vx + vy * vy + vz * vz);
  }
}

__global__ void degree_kernel(const int* __restrict__ rowI, float* __restrict__ cnt, int E) {
  const int e = blockIdx.x * blockDim.x + threadIdx.x;
  if (e < E) atomicAdd(&cnt[rowI[e]], 1.0f);
}

__global__ void zero_kernel(float* __restrict__ p, int n) {
  const int i = blockIdx.x * blockDim.x + threadIdx.x;
  if (i < n) p[i] = 0.0f;
}

extern "C" void kernel_launch(void* const* d_in, const int* in_sizes, int n_in,
                              void* d_out, int out_size, void* d_ws, size_t ws_size,
                              hipStream_t stream) {
  const float* his  = (const float*)d_in[0];
  const float* loc  = (const float*)d_in[1];
  const int*  edges = (const int*)  d_in[2];
  const float* vel  = (const float*)d_in[3];
  const float* eatt = (const float*)d_in[4];
  const float* embW = (const float*)d_in[5];
  const float* embB = (const float*)d_in[6];
  const float* eW1  = (const float*)d_in[7];
  const float* eb1  = (const float*)d_in[8];
  const float* eW2  = (const float*)d_in[9];
  const float* eb2  = (const float*)d_in[10];
  const float* cW1  = (const float*)d_in[11];
  const float* cb1  = (const float*)d_in[12];
  const float* cW2  = (const float*)d_in[13];
  const float* vW1  = (const float*)d_in[14];
  const float* vb1  = (const float*)d_in[15];
  const float* vW2  = (const float*)d_in[16];
  const float* vb2  = (const float*)d_in[17];
  const float* nW1  = (const float*)d_in[18];
  const float* nb1  = (const float*)d_in[19];
  const float* nW2  = (const float*)d_in[20];
  const float* nb2  = (const float*)d_in[21];

  const int N = in_sizes[0] / 8;      // his is [N,8]
  const int E = in_sizes[2] / 2;      // edges is [2,E]
  const int* rowI = edges;
  const int* colI = edges + E;

  // outputs live directly in d_out: x [N,3] | h [N,64] | v [N,3]
  float* out = (float*)d_out;
  float* x = out;
  float* h = out + (size_t)N * 3;
  float* v = out + (size_t)N * 67;

  // workspace: speed0 [N] | cnt [N] | acc3 [N,3] | agg_h [N,64] | h16 [N,64]f16
  float* ws   = (float*)d_ws;
  float* sp0  = ws;
  float* cnt  = ws + N;
  float* acc3 = ws + (size_t)2 * N;
  float* aggh = ws + (size_t)5 * N;
  _Float16* h16 = (_Float16*)(ws + (size_t)69 * N);

  zero_kernel<<<(N + 255) / 256, 256, 0, stream>>>(cnt, N);
  embed_kernel<<<(N + 255) / 256, 256, 0, stream>>>(his, loc, vel, embW, embB,
                                                    x, h, h16, v, sp0, N);
  degree_kernel<<<(E + 255) / 256, 256, 0, stream>>>(rowI, cnt, E);

  for (int l = 0; l < 4; ++l) {
    zero_kernel<<<((N * 67) + 255) / 256, 256, 0, stream>>>(acc3, N * 67);
    edge_kernel<<<1280, EWAVES * 32, 0, stream>>>(h16, x, rowI, colI, eatt,
        eW1, eb1, eW2, eb2, cW1, cb1, cW2, acc3, aggh, E);
    vel_kernel<<<(N + 255) / 256, 256, 0, stream>>>(h, acc3, cnt, sp0, x, v,
        vW1, vb1, vW2, vb2, N);
    nodeh_kernel<<<2500, 256, 0, stream>>>(h, h16, aggh, nW1, nb1, nW2, nb2, N);
  }
}
